// CRFLoss_vb_71631464563253
// MI455X (gfx1250) — matrix operand order repaired and checked by reading, not compiled, not run on hardware
//
#include <hip/hip_runtime.h>
#include <stdint.h>

typedef __attribute__((ext_vector_type(16))) _Float16 v16h;
typedef __attribute__((ext_vector_type(8)))  float    v8f;
typedef __attribute__((ext_vector_type(4)))  unsigned int v4u;
typedef __attribute__((ext_vector_type(8)))  int      v8i;
typedef __attribute__((ext_vector_type(4)))  int      v4i;

#define TAGSET   64
#define SEQ_LEN  512
#define BATCH    64
#define START_TAG 62
#define END_TAG   63
#define ROWP     66   // padded LDS row stride in floats (64 data + 2 pad dwords via TDM pad)

// Issue a TDM load of one 64x64 fp32 tile (row-major, contiguous) into LDS,
// padding +2 dwords every 64 dwords so LDS rows land at stride 66 (conflict-free
// column walks for the WMMA B operand). D# per CDNA5 ISA.
__device__ __forceinline__ void tdm_load_tile(const float* gsrc, uint32_t lds_off) {
  uint64_t ga = (uint64_t)(uintptr_t)gsrc;
  v4u g0;
  g0.x = 1u;                                   // count=1, user descriptor
  g0.y = lds_off;                              // lds_addr [63:32]
  g0.z = (uint32_t)ga;                         // global_addr lo
  g0.w = (uint32_t)(ga >> 32) | 0x80000000u;   // global_addr hi [56:32] | type=2 @ [127:126]
  v8i g1;
  g1[0] = (2 << 16)        // data_size = 4B
        | (1 << 20)        // pad_enable
        | (5 << 22)        // pad_interval: 64 dwords
        | (1 << 25);       // pad_amount: 2 dwords
  g1[1] = (TAGSET << 16);  // tensor_dim0 = 64 (low 16 bits at [63:48])
  g1[2] = (TAGSET << 16);  // tensor_dim0 hi = 0, tensor_dim1 = 64 (low 16)
  g1[3] = (TAGSET << 16);  // tensor_dim1 hi = 0, tile_dim0 = 64
  g1[4] = TAGSET;          // tile_dim1 = 64, tile_dim2 = 0
  g1[5] = TAGSET;          // tensor_dim0_stride = 64 elements (lo 32)
  g1[6] = 0;               // stride hi / dim1_stride (unused for 2D tile)
  g1[7] = 0;
  v4i z4 = {0, 0, 0, 0};
  v8i z8 = {0, 0, 0, 0, 0, 0, 0, 0};
  __builtin_amdgcn_tensor_load_to_lds(g0, g1, z4, z4, z8, 0);
}

__global__ __launch_bounds__(128) void crf_loss_kernel(
    const float* __restrict__ scores, const int* __restrict__ target,
    const int* __restrict__ mask, float* __restrict__ out) {
  __shared__ float    tiles[3][TAGSET * ROWP];
  __shared__ float    curb[TAGSET];
  __shared__ float    partb[TAGSET];
  __shared__ _Float16 wb[TAGSET];
  __shared__ float    Mv;

  const int b    = blockIdx.x;
  const int tid  = threadIdx.x;
  const int wid  = tid >> 5;           // 4 waves, one per 16-wide j tile
  const int lane = tid & 31;
  const int jb   = wid << 4;
  const int n    = lane & 15;
  const int koff = (lane >= 16) ? 16 : 0;  // B layout: upper half-wave holds K+16
  const int aoff = (lane >= 16) ? 8  : 0;  // unused: A is loaded with koff, upper half-wave holds K+16

  float numer = 0.0f;

  // Prologue: prefetch tiles 0..2 into the LDS ring, wait for tile 0.
  if (tid == 0) {
    tdm_load_tile(scores + (size_t)b * 4096,                 (uint32_t)(uintptr_t)&tiles[0][0]);
    tdm_load_tile(scores + ((size_t)1 * BATCH + b) * 4096,   (uint32_t)(uintptr_t)&tiles[1][0]);
    tdm_load_tile(scores + ((size_t)2 * BATCH + b) * 4096,   (uint32_t)(uintptr_t)&tiles[2][0]);
    __builtin_amdgcn_s_wait_tensorcnt(2);
  }
  __syncthreads();

  // ini = scores[0][:, START_TAG, :]
  if (tid < TAGSET) {
    float v = tiles[0][START_TAG * ROWP + tid];
    curb[tid]  = v;
    partb[tid] = v;
  }
  if (tid == 0) {
    int tg = target[b];
    numer += tiles[0][(tg >> 6) * ROWP + (tg & 63)] * (float)mask[b];
  }
  __syncthreads();

  for (int t = 1; t < SEQ_LEN; ++t) {
    // Tile t ready (loads complete in order; one load still in flight except at the end).
    if (tid == 0) {
      if (t == SEQ_LEN - 1) __builtin_amdgcn_s_wait_tensorcnt(0);
      else                  __builtin_amdgcn_s_wait_tensorcnt(1);
    }
    __syncthreads();  // everyone sees tile t; prev iteration's reads of ring slot (t+2)%3 done

    if (tid == 0 && (t + 2) < SEQ_LEN) {
      tdm_load_tile(scores + ((size_t)(t + 2) * BATCH + b) * 4096,
                    (uint32_t)(uintptr_t)&tiles[(t + 2) % 3][0]);
    }

    const float* tile = tiles[t % 3];
    const int mt = mask[t * BATCH + b];

    // Phase 1 (wave 0): M = max_i cur[i]; w[i] = exp(cur[i]-M) -> f16; numerator gather.
    if (wid == 0) {
      float c0 = curb[lane];
      float c1 = curb[lane + 32];
      float m  = fmaxf(c0, c1);
      #pragma unroll
      for (int s = 16; s >= 1; s >>= 1) m = fmaxf(m, __shfl_xor(m, s, 32));
      wb[lane]      = (_Float16)__expf(c0 - m);
      wb[lane + 32] = (_Float16)__expf(c1 - m);
      if (lane == 0) {
        Mv = m;
        int tg = target[t * BATCH + b];
        numer += tile[(tg >> 6) * ROWP + (tg & 63)] * (float)mt;
      }
    }
    __syncthreads();  // wb, Mv visible

    // Phase 2 (all 4 waves): r[j] = sum_i exp(S[i][j]) * w[i] via two chained
    // v_wmma_f32_16x16x32_f16 (K=64). A = w replicated across all 16 rows.
    v16h A0, A1, B0, B1;
    #pragma unroll
    for (int h = 0; h < 8; ++h) {
      A0[h]     = wb[koff + h];            // K = 0..7   (+16 for upper half-wave)
      A0[8 + h] = wb[8 + koff + h];        // K = 8..15  (+16)
      A1[h]     = wb[32 + koff + h];       // K = 32..39 (+16)
      A1[8 + h] = wb[40 + koff + h];       // K = 40..47 (+16)
    }
    #pragma unroll
    for (int h = 0; h < 16; ++h) {
      B0[h] = (_Float16)__expf(tile[(koff + h) * ROWP + jb + n]);        // K = h (+16)
      B1[h] = (_Float16)__expf(tile[(32 + koff + h) * ROWP + jb + n]);   // K = 32+h (+16)
    }
    v8f c = {};
    c = __builtin_amdgcn_wmma_f32_16x16x32_f16(false, A0, false, B0, (short)0, c, false, false);
    c = __builtin_amdgcn_wmma_f32_16x16x32_f16(false, A1, false, B1, (short)0, c, false, false);

    // All D rows are identical (A rows replicated); lanes 0..15 hold r[jb+n] in c[0].
    if (lane < 16) {
      float nc = Mv + __logf(c[0]);
      int j = jb + lane;
      curb[j] = nc;                 // cur_partition always updated
      if (mt) partb[j] = nc;        // partition updated only where mask > 0
    }
  }

  __syncthreads();
  if (tid == 0) {
    out[b] = (partb[END_TAG] - numer) * (1.0f / (float)BATCH);
  }
}

extern "C" void kernel_launch(void* const* d_in, const int* in_sizes, int n_in,
                              void* d_out, int out_size, void* d_ws, size_t ws_size,
                              hipStream_t stream) {
  const float* scores = (const float*)d_in[0];
  const int*   target = (const int*)d_in[1];
  const int*   mask   = (const int*)d_in[2];
  float*       out    = (float*)d_out;
  crf_loss_kernel<<<dim3(BATCH), dim3(128), 0, stream>>>(scores, target, mask, out);
}
